// ScalableQuantumTransformer_18940805775805
// MI455X (gfx1250) — compile-verified, matrix operand order checked by reading, not run on hardware
//
#include <hip/hip_runtime.h>
#include <hip/hip_bf16.h>

// ---------------------------------------------------------------------------
// Types for CDNA5 WMMA (wave32, 16x16x32 bf16 -> f32)
// ---------------------------------------------------------------------------
typedef __attribute__((ext_vector_type(16))) unsigned short v16u16;
typedef __attribute__((ext_vector_type(8)))  unsigned short v8u16;
typedef __attribute__((ext_vector_type(16))) __bf16        v16bf;
typedef __attribute__((ext_vector_type(8)))  float         v8f;
typedef __attribute__((ext_vector_type(4)))  int           v4i;
typedef __attribute__((address_space(1)))    v4i           gas_v4i;  // global (AS1)
typedef __attribute__((address_space(3)))    v4i           las_v4i;  // LDS (AS3)

__device__ inline unsigned short f2bf_u16(float f) {
    union { float f; unsigned int u; } x; x.f = f;
    unsigned int u = x.u;
    unsigned int r = u + 0x7FFFu + ((u >> 16) & 1u);           // round-nearest-even
    r = ((u & 0x7F800000u) == 0x7F800000u) ? u : r;            // inf/nan passthrough
    return (unsigned short)(r >> 16);
}

// ---------------------------------------------------------------------------
// Async global -> LDS copy (gfx1250 GLOBAL_LOAD_ASYNC_TO_LDS_B128, ASYNCcnt),
// with a synchronous fallback if this toolchain lacks the builtin.
// ---------------------------------------------------------------------------
#if __has_builtin(__builtin_amdgcn_global_load_async_to_lds_b128)
#define HAVE_ASYNC_LDS 1
#else
#define HAVE_ASYNC_LDS 0
#endif

__device__ inline void cp16_g2l(const unsigned short* g, unsigned short* l) {
#if HAVE_ASYNC_LDS
    // address-space casts must go through integers in HIP C++;
    // low 32 bits of a generic __shared__ pointer are the LDS byte address.
    gas_v4i* gp = (gas_v4i*)(unsigned long long)g;
    las_v4i* lp = (las_v4i*)(unsigned int)(unsigned long long)l;
    __builtin_amdgcn_global_load_async_to_lds_b128(gp, lp, 0, 0);
#else
    *(v8u16*)l = *(const v8u16*)g;
#endif
}

__device__ inline void async_wait0() {
#if HAVE_ASYNC_LDS
#if __has_builtin(__builtin_amdgcn_s_wait_asynccnt)
    __builtin_amdgcn_s_wait_asynccnt(0);
#else
    asm volatile("s_wait_asynccnt 0x0" ::: "memory");
#endif
#endif
}

// Load one 16x32 bf16 fragment (A-layout) for a row-major matrix whose
// fastest dimension is K. Per the CDNA5 ISA layout, lane l holds row (l&15)
// and two contiguous 8-element K chunks at kbase=(l>>4)*8 and 16+kbase,
// i.e. exactly two 16-byte b128 loads per lane. The identical function
// serves B fragments for any K-contiguous [N,K] matrix (row index = n).
__device__ inline v16u16 ldfrag(const unsigned short* __restrict__ p, int ld, int lane) {
    int r  = lane & 15;
    int kb = (lane >> 4) << 3;
    const unsigned short* row = p + (size_t)r * ld;
    v8u16 lo = *(const v8u16*)(row + kb);
    v8u16 hi = *(const v8u16*)(row + 16 + kb);
    return __builtin_shufflevector(lo, hi, 0,1,2,3,4,5,6,7,8,9,10,11,12,13,14,15);
}

// Same fragment gather from an LDS panel with ld = 32 elements (ds_load_b128 x2).
__device__ inline v16u16 ldfrag_lds(const unsigned short* base, int lane) {
    int r  = lane & 15;
    int kb = (lane >> 4) << 3;
    const unsigned short* row = base + r * 32;
    v8u16 lo = *(const v8u16*)(row + kb);
    v8u16 hi = *(const v8u16*)(row + 16 + kb);
    return __builtin_shufflevector(lo, hi, 0,1,2,3,4,5,6,7,8,9,10,11,12,13,14,15);
}

__device__ inline v8f wmma_bf16(v16u16 a, v16u16 b, v8f c) {
    union { v16u16 u; v16bf bf; } ua, ub;
    ua.u = a; ub.u = b;
    return __builtin_amdgcn_wmma_f32_16x16x32_bf16(
        /*neg_a=*/false, ua.bf, /*neg_b=*/false, ub.bf,
        /*c_mod=*/(short)0, c, /*reuse_a=*/false, /*reuse_b=*/false);
}

// ---------------------------------------------------------------------------
// fp32 -> bf16 converter (grid-stride)
// ---------------------------------------------------------------------------
__global__ __launch_bounds__(256) void cvt_f32_bf16_kernel(
    const float* __restrict__ in, unsigned short* __restrict__ out, size_t n) {
    size_t i = (size_t)blockIdx.x * 256 + threadIdx.x;
    size_t stride = (size_t)gridDim.x * 256;
    for (; i < n; i += stride) out[i] = f2bf_u16(in[i]);
}

// ---------------------------------------------------------------------------
// bf16 WMMA GEMM:  out[m,n] = sum_k A[m,k] * W[n,k]   (+bias, +resid, relu)
// Block = 256 threads (8 waves) -> 128x128 tile; wave tile 32x64.
// A/B 128x32 panels staged in LDS via async global->LDS DMA, double-buffered
// so panel i+1 streams in (ASYNCcnt) while panel i feeds the WMMAs.
// M,N multiples of 128; K multiple of 32 (true for all call sites).
// ---------------------------------------------------------------------------
__global__ __launch_bounds__(256) void gemm_bf16_kernel(
    const unsigned short* __restrict__ A, int lda,
    const unsigned short* __restrict__ W, int ldw,
    const float* __restrict__ bias,
    const float* __restrict__ resid,
    float* __restrict__ outF,
    unsigned short* __restrict__ outB,
    int N, int K, int relu) {
    __shared__ unsigned short sA[2][128 * 32];   // 8 KB per buffer
    __shared__ unsigned short sB[2][128 * 32];   // 32 KB total

    int tid  = threadIdx.x;
    int lane = tid & 31;
    int wave = tid >> 5;
    int wm = wave & 3, wn = wave >> 2;                 // 4 (M) x 2 (N) wave grid
    int mBase = blockIdx.y * 128;
    int nBase = blockIdx.x * 128;
    const unsigned short* Ag = A + (size_t)mBase * lda;
    const unsigned short* Wg = W + (size_t)nBase * ldw;

    // each thread DMAs two 16B chunks per panel per K-step
    int crow = tid >> 1;            // 0..127
    int ccol = (tid & 1) << 3;      // 0 or 8 (elements)
    auto stage = [&](int buf, int k0) {
        const unsigned short* ga = Ag + (size_t)crow * lda + k0 + ccol;
        cp16_g2l(ga,      &sA[buf][crow * 32 + ccol]);
        cp16_g2l(ga + 16, &sA[buf][crow * 32 + ccol + 16]);
        const unsigned short* gw = Wg + (size_t)crow * ldw + k0 + ccol;
        cp16_g2l(gw,      &sB[buf][crow * 32 + ccol]);
        cp16_g2l(gw + 16, &sB[buf][crow * 32 + ccol + 16]);
    };

    v8f acc[2][4] = {};
    stage(0, 0);
    int nbuf = 1;
    for (int k0 = 0; k0 < K; k0 += 32) {
        int cur = nbuf ^ 1;
        async_wait0();               // this wave's DMA into `cur` complete
        __syncthreads();             // => all waves' DMA complete, prev compute done
        if (k0 + 32 < K) stage(nbuf, k0 + 32);   // overlap next panel DMA
        v16u16 a[2], b[4];
#pragma unroll
        for (int mt = 0; mt < 2; mt++)
            a[mt] = ldfrag_lds(&sA[cur][(wm * 32 + mt * 16) * 32], lane);
#pragma unroll
        for (int nt = 0; nt < 4; nt++)
            b[nt] = ldfrag_lds(&sB[cur][(wn * 64 + nt * 16) * 32], lane);
#pragma unroll
        for (int mt = 0; mt < 2; mt++)
#pragma unroll
            for (int nt = 0; nt < 4; nt++)
                acc[mt][nt] = wmma_bf16(a[mt], b[nt], acc[mt][nt]);
        nbuf ^= 1;
    }

    int cl = lane & 15;
    int rowOff = (lane >> 4) * 8;
#pragma unroll
    for (int mt = 0; mt < 2; mt++) {
#pragma unroll
        for (int nt = 0; nt < 4; nt++) {
            int col = nBase + wn * 64 + nt * 16 + cl;
            float bv = bias ? bias[col] : 0.f;
#pragma unroll
            for (int r = 0; r < 8; r++) {
                int row = mBase + wm * 32 + mt * 16 + rowOff + r;
                size_t idx = (size_t)row * N + col;
                float v = acc[mt][nt][r] + bv;
                if (resid) v += resid[idx];
                if (relu)  v = fmaxf(v, 0.f);
                if (outF)  outF[idx] = v;
                if (outB)  outB[idx] = f2bf_u16(v);
            }
        }
    }
}

// ---------------------------------------------------------------------------
// Build V^T:  vt[(b*H+h)*dh*S + d*S + s] = qkv[(b*S+s)*3D + 2D + h*dh + d]
// (gives key-contiguous B fragments for the P x V WMMA)
// ---------------------------------------------------------------------------
__global__ __launch_bounds__(256) void build_vt_kernel(
    const unsigned short* __restrict__ qkv, unsigned short* __restrict__ vt) {
    const int S = 4096, H = 12, QKV = 2304;
    size_t id = (size_t)blockIdx.x * 256 + threadIdx.x;
    const size_t total = (size_t)2 * H * 64 * S;
    if (id >= total) return;
    int s  = (int)(id & (S - 1));
    int d  = (int)((id >> 12) & 63);
    int bh = (int)(id >> 18);
    int b = bh / H, h = bh - b * H;
    vt[id] = qkv[(size_t)(b * S + s) * QKV + 1536 + h * 64 + d];
}

// ---------------------------------------------------------------------------
// Flash attention (bf16 WMMA core, f32 online softmax).
// grid = (S/64, B*H); 128 threads = 4 waves; each wave owns 16 q-rows.
// ---------------------------------------------------------------------------
__global__ __launch_bounds__(128) void attention_kernel(
    const unsigned short* __restrict__ qkv,
    const unsigned short* __restrict__ vt,
    unsigned short* __restrict__ ctx) {
    const int S = 4096, QKV = 2304, H = 12, dh = 64, Dm = 768;
    int bh = blockIdx.y;
    int b = bh / H, h = bh - b * H;
    int lane = threadIdx.x & 31, wave = threadIdx.x >> 5;
    int qtok = b * S + blockIdx.x * 64 + wave * 16;       // absolute token of row 0

    const unsigned short* Qp = qkv + (size_t)qtok * QKV + h * dh;
    v16u16 qa0 = ldfrag(Qp,      QKV, lane);              // K-dim 0..31 of dh
    v16u16 qa1 = ldfrag(Qp + 32, QKV, lane);              // K-dim 32..63
    const unsigned short* Kp = qkv + (size_t)(b * S) * QKV + Dm + h * dh;
    const unsigned short* Vp = vt + (size_t)bh * dh * S;

    __shared__ unsigned short pl[4][512];                 // per-wave 16x32 bf16 P tile

    float mi[8], li[8];
#pragma unroll
    for (int r = 0; r < 8; r++) { mi[r] = -1e30f; li[r] = 0.f; }
    v8f o0 = {}, o1 = {}, o2 = {}, o3 = {};
    const float sc = 0.125f;                              // 1/sqrt(64)
    int rowOff = (lane >> 4) * 8;
    int cl = lane & 15;

    for (int kb = 0; kb < S; kb += 32) {
        // ---- scores: two 16x16 tiles over 32 keys, K-dim = dh = 64 ----
        v8f s0 = {}, s1 = {};
        {
            v16u16 kf0 = ldfrag(Kp + (size_t)kb * QKV,             QKV, lane);
            v16u16 kf1 = ldfrag(Kp + (size_t)kb * QKV + 32,        QKV, lane);
            s0 = wmma_bf16(qa0, kf0, s0);
            s0 = wmma_bf16(qa1, kf1, s0);
            v16u16 kf2 = ldfrag(Kp + (size_t)(kb + 16) * QKV,      QKV, lane);
            v16u16 kf3 = ldfrag(Kp + (size_t)(kb + 16) * QKV + 32, QKV, lane);
            s1 = wmma_bf16(qa0, kf2, s1);
            s1 = wmma_bf16(qa1, kf3, s1);
        }
        // ---- online softmax (row reductions stay within 16-lane halves) ----
#pragma unroll
        for (int r = 0; r < 8; r++) {
            float a  = s0[r] * sc, bb = s1[r] * sc;
            float mx = fmaxf(a, bb);
#pragma unroll
            for (int m = 8; m; m >>= 1) mx = fmaxf(mx, __shfl_xor(mx, m, 32));
            float mnew = fmaxf(mi[r], mx);
            float corr = __expf(mi[r] - mnew);
            mi[r] = mnew;
            float p0 = __expf(a  - mnew);
            float p1 = __expf(bb - mnew);
            float ps = p0 + p1;
#pragma unroll
            for (int m = 8; m; m >>= 1) ps += __shfl_xor(ps, m, 32);
            li[r] = li[r] * corr + ps;
            o0[r] *= corr; o1[r] *= corr; o2[r] *= corr; o3[r] *= corr;
            int mrow = rowOff + r;
            pl[wave][mrow * 32 + cl]      = f2bf_u16(p0);
            pl[wave][mrow * 32 + 16 + cl] = f2bf_u16(p1);
        }
        __builtin_amdgcn_wave_barrier();                  // order DS store -> DS load
        // ---- re-read P as an A fragment (16x32 bf16) from LDS ----
        v16u16 pf;
        {
            int kb2 = (lane >> 4) * 8;
            v8u16 lo = *(const v8u16*)&pl[wave][cl * 32 + kb2];
            v8u16 hi = *(const v8u16*)&pl[wave][cl * 32 + 16 + kb2];
            pf = __builtin_shufflevector(lo, hi, 0,1,2,3,4,5,6,7,8,9,10,11,12,13,14,15);
        }
        // ---- O += P (16x32) x V (32x64): four n-tiles from V^T ----
        v16u16 vf0 = ldfrag(Vp + (size_t)0  * S + kb, S, lane);
        v16u16 vf1 = ldfrag(Vp + (size_t)16 * S + kb, S, lane);
        v16u16 vf2 = ldfrag(Vp + (size_t)32 * S + kb, S, lane);
        v16u16 vf3 = ldfrag(Vp + (size_t)48 * S + kb, S, lane);
        o0 = wmma_bf16(pf, vf0, o0);
        o1 = wmma_bf16(pf, vf1, o1);
        o2 = wmma_bf16(pf, vf2, o2);
        o3 = wmma_bf16(pf, vf3, o3);
    }
#pragma unroll
    for (int r = 0; r < 8; r++) {
        float inv = 1.f / li[r];
        size_t tok = (size_t)(qtok + rowOff + r);
        unsigned short* cp = ctx + tok * Dm + h * dh + cl;
        cp[0]  = f2bf_u16(o0[r] * inv);
        cp[16] = f2bf_u16(o1[r] * inv);
        cp[32] = f2bf_u16(o2[r] * inv);
        cp[48] = f2bf_u16(o3[r] * inv);
    }
}

// ---------------------------------------------------------------------------
// LayerNorm over D=768; one block (256 thr) per row. Optional bf16 copy.
// ---------------------------------------------------------------------------
__global__ __launch_bounds__(256) void layernorm_kernel(
    const float* __restrict__ in, const float* __restrict__ w, const float* __restrict__ b,
    float* __restrict__ outF, unsigned short* __restrict__ outB) {
    const int D = 768;
    int row = blockIdx.x;
    int lane = threadIdx.x & 31, wave = threadIdx.x >> 5;
    const float* x = in + (size_t)row * D;
    float s = 0.f, s2 = 0.f;
#pragma unroll
    for (int i = threadIdx.x; i < D; i += 256) { float v = x[i]; s += v; s2 += v * v; }
#pragma unroll
    for (int m = 16; m; m >>= 1) { s += __shfl_xor(s, m, 32); s2 += __shfl_xor(s2, m, 32); }
    __shared__ float rs[8], rs2[8];
    if (lane == 0) { rs[wave] = s; rs2[wave] = s2; }
    __syncthreads();
    s = 0.f; s2 = 0.f;
#pragma unroll
    for (int i = 0; i < 8; i++) { s += rs[i]; s2 += rs2[i]; }
    float mu   = s * (1.f / D);
    float var  = s2 * (1.f / D) - mu * mu;
    float rstd = rsqrtf(var + 1e-5f);
#pragma unroll
    for (int i = threadIdx.x; i < D; i += 256) {
        float v = (x[i] - mu) * rstd * w[i] + b[i];
        if (outF) outF[(size_t)row * D + i] = v;
        if (outB) outB[(size_t)row * D + i] = f2bf_u16(v);
    }
}

// ---------------------------------------------------------------------------
// Launch: QKV GEMM -> V^T -> flash attention -> out-proj (+src, LN1)
//         -> FFN1 (relu) -> FFN2 (+x) -> LN2 -> d_out
// ---------------------------------------------------------------------------
extern "C" void kernel_launch(void* const* d_in, const int* in_sizes, int n_in,
                              void* d_out, int out_size, void* d_ws, size_t ws_size,
                              hipStream_t stream) {
    (void)in_sizes; (void)n_in; (void)out_size; (void)ws_size;
    const float* src   = (const float*)d_in[0];
    const float* w_qkv = (const float*)d_in[1];
    const float* b_qkv = (const float*)d_in[2];
    const float* w_out = (const float*)d_in[3];
    const float* b_out = (const float*)d_in[4];
    const float* w1    = (const float*)d_in[5];
    const float* b1    = (const float*)d_in[6];
    const float* w2    = (const float*)d_in[7];
    const float* b2    = (const float*)d_in[8];
    const float* ln1w  = (const float*)d_in[9];
    const float* ln1b  = (const float*)d_in[10];
    const float* ln2w  = (const float*)d_in[11];
    const float* ln2b  = (const float*)d_in[12];

    const int S = 4096, D = 768, DFF = 3072, H = 12;
    const int T = 2 * S;                      // 8192 tokens
    const int QKV = 3 * D;                    // 2304

    char* p = (char*)d_ws;
    auto alloc = [&](size_t n) { char* q = p; p += (n + 255) & ~(size_t)255; return q; };
    unsigned short* Xbf    = (unsigned short*)alloc((size_t)T * D * 2);
    unsigned short* Wqkvbf = (unsigned short*)alloc((size_t)QKV * D * 2);
    unsigned short* Woutbf = (unsigned short*)alloc((size_t)D * D * 2);
    unsigned short* W1bf   = (unsigned short*)alloc((size_t)DFF * D * 2);
    unsigned short* W2bf   = (unsigned short*)alloc((size_t)D * DFF * 2);
    unsigned short* qkvbf  = (unsigned short*)alloc((size_t)T * QKV * 2);
    unsigned short* vtbf   = (unsigned short*)alloc((size_t)2 * H * 64 * S * 2);
    unsigned short* ctxbf  = (unsigned short*)alloc((size_t)T * D * 2);
    float*          ybuf   = (float*)alloc((size_t)T * D * 4);   // attn-out + src; reused for ffn + x
    float*          xf32   = (float*)alloc((size_t)T * D * 4);   // LN1 output
    unsigned short* xbf    = (unsigned short*)alloc((size_t)T * D * 2);
    unsigned short* hbf    = (unsigned short*)alloc((size_t)T * DFF * 2);

    auto cvt = [&](const float* in, unsigned short* out, size_t n) {
        int grid = (int)((n + 255) / 256); if (grid > 4096) grid = 4096;
        cvt_f32_bf16_kernel<<<grid, 256, 0, stream>>>(in, out, n);
    };
    cvt(src,   Xbf,    (size_t)T * D);
    cvt(w_qkv, Wqkvbf, (size_t)QKV * D);
    cvt(w_out, Woutbf, (size_t)D * D);
    cvt(w1,    W1bf,   (size_t)DFF * D);
    cvt(w2,    W2bf,   (size_t)D * DFF);

    // QKV projection: [T,D] x [QKV,D]^T -> bf16 [T,QKV]
    gemm_bf16_kernel<<<dim3(QKV / 128, T / 128), 256, 0, stream>>>(
        Xbf, D, Wqkvbf, D, b_qkv, nullptr, nullptr, qkvbf, QKV, D, 0);

    // V^T for key-contiguous PV fragments
    {
        size_t n = (size_t)2 * H * 64 * S;
        build_vt_kernel<<<(int)((n + 255) / 256), 256, 0, stream>>>(qkvbf, vtbf);
    }

    // flash attention -> ctx bf16 [T,D]
    attention_kernel<<<dim3(S / 64, 2 * H), 128, 0, stream>>>(qkvbf, vtbf, ctxbf);

    // out-projection + b_out + src residual -> ybuf f32
    gemm_bf16_kernel<<<dim3(D / 128, T / 128), 256, 0, stream>>>(
        ctxbf, D, Woutbf, D, b_out, src, ybuf, nullptr, D, D, 0);

    // LN1 -> xf32 (+ bf16 copy for FFN input)
    layernorm_kernel<<<T, 256, 0, stream>>>(ybuf, ln1w, ln1b, xf32, xbf);

    // FFN1: relu(x @ W1^T + b1) -> bf16 [T,DFF]
    gemm_bf16_kernel<<<dim3(DFF / 128, T / 128), 256, 0, stream>>>(
        xbf, D, W1bf, D, b1, nullptr, nullptr, hbf, DFF, D, 1);

    // FFN2: h @ W2^T + b2 + x residual -> ybuf f32 (reuse)
    gemm_bf16_kernel<<<dim3(D / 128, T / 128), 256, 0, stream>>>(
        hbf, DFF, W2bf, DFF, b2, xf32, ybuf, nullptr, D, DFF, 0);

    // LN2 -> d_out f32
    layernorm_kernel<<<T, 256, 0, stream>>>(ybuf, ln2w, ln2b, (float*)d_out, nullptr);
}
